// CompatV1LSTM_11879879543021
// MI455X (gfx1250) — matrix operand order repaired and checked by reading, not compiled, not run on hardware
//
#include <hip/hip_runtime.h>

// CDNA5 / gfx1250 fused LSTM. B=512, T=128, D=256, H=256, C=128, 4H=1024.
//
//  - prep: Wt = W^T (bf16, [1024][512]), WoutT = w_out^T (bf16, [128][256]);
//          zero c, h0. Transposed weights stay L2-resident for all 128 steps.
//  - 128x lstm_step (sequential recurrence, h double-buffered, bf16 WMMA,
//          fp32 accum, gates fused in epilogue). grid (16,8) x 128 threads.
//          Global->VGPR staging of chunk k+1 overlaps WMMAs of chunk k
//          (register software pipeline, fully unrolled K loop).
//  - out_proj: out = h_last @ w_out + b_out via WMMA.
//
// All global staging is 128-bit; LDS rows use an 80-element bf16 pitch
// (160B, multiple of 16B) so every ds_*_b128 is naturally aligned.

typedef __attribute__((ext_vector_type(16))) __bf16 v16bf;
typedef __attribute__((ext_vector_type(8)))  __bf16 bf16x8;
typedef __attribute__((ext_vector_type(4)))  __bf16 bf16x4;
typedef __attribute__((ext_vector_type(4)))  float  f32x4;
typedef __attribute__((ext_vector_type(8)))  float  v8f;

#define B_  512
#define T_  128
#define D_  256
#define H_  256
#define C_  128
#define G4H 1024
#define LP  80   // LDS bf16 pitch (160 bytes)

__device__ __forceinline__ float sigf(float x) {
    return 1.0f / (1.0f + __expf(-x));
}

// A-fragment, CDNA5 16-bit A 16x32 layout:
// lane L: M=L%16; K(e) = (e&7) + (e>=8?16:0) + (L>=16?8:0)
__device__ __forceinline__ v16bf load_a_frag(const __bf16* rowp, int kbase) {
    bf16x8 lo = *(const bf16x8*)(rowp + kbase);
    bf16x8 hi = *(const bf16x8*)(rowp + kbase + 16);
    return __builtin_shufflevector(lo, hi, 0,1,2,3,4,5,6,7,8,9,10,11,12,13,14,15);
}

// B-fragment, CDNA5 16-bit B 32x16 layout (LDS holds B as [n][k]):
// lane L: N=L%16; K(e) = (e&7) + (e>=8?16:0) + (L>=16?8:0)
__device__ __forceinline__ v16bf load_b_frag(const __bf16* colp, int kbase) {
    bf16x8 lo = *(const bf16x8*)(colp + kbase);
    bf16x8 hi = *(const bf16x8*)(colp + kbase + 16);
    return __builtin_shufflevector(lo, hi, 0,1,2,3,4,5,6,7,8,9,10,11,12,13,14,15);
}

// ---------------------------------------------------------------------------
// Prep: Wt = W^T bf16, WoutT = w_out^T bf16, zero c / h0.
// ---------------------------------------------------------------------------
__global__ void lstm_prep_kernel(const float* __restrict__ W,
                                 const float* __restrict__ w_out,
                                 __bf16* __restrict__ Wt,     // [4H][D+H]
                                 __bf16* __restrict__ WoutT,  // [C][H]
                                 float* __restrict__ c,
                                 __bf16* __restrict__ h0) {
    int i = blockIdx.x * blockDim.x + threadIdx.x;
    if (i < G4H * (D_ + H_)) {
        int col = i >> 9, k = i & 511;           // Wt[col][k] = W[k][col]
        Wt[i] = (__bf16)W[(size_t)k * G4H + col];
    }
    if (i < C_ * H_) {
        int n = i >> 8, k = i & 255;             // WoutT[n][k] = w_out[k][n]
        WoutT[i] = (__bf16)w_out[(size_t)k * C_ + n];
    }
    if (i < B_ * H_) {
        c[i]  = 0.0f;
        h0[i] = (__bf16)0.0f;
    }
}

// ---------------------------------------------------------------------------
// One LSTM timestep. grid (16,8): 32-row batch tile x 32-col H tile.
// block = 128 threads = 4 wave32s (2 M-slabs x 2 N-slabs).
// Each wave: one 16x16 tile for each of the 4 gates -> 4 v8f accumulators.
// ---------------------------------------------------------------------------
__global__ void __launch_bounds__(128)
lstm_step_kernel(const float* __restrict__ x,      // [B,T,D]
                 const __bf16* __restrict__ Wt,    // [4H][D+H] (transposed)
                 const float* __restrict__ bias,   // [4H]
                 const __bf16* __restrict__ h_in,  // [B,H]
                 __bf16* __restrict__ h_out,       // [B,H]
                 float* __restrict__ c,            // [B,H] in/out
                 int t) {
    __shared__ __bf16 As[32][LP];        // [row][k]
    __shared__ __bf16 Bs[4][32][LP];     // [gate][n][k]

    const int tid  = threadIdx.x;
    const int lane = tid & 31;
    const int wave = tid >> 5;
    const int wr   = wave >> 1;          // 0..1: 16-row slab
    const int wc   = wave & 1;           // 0..1: 16-col slab
    const int bm0  = blockIdx.x * 32;    // batch tile base
    const int bh0  = blockIdx.y * 32;    // H tile base
    const int lhi  = (lane & 16) ? 1 : 0;

    // Prefetch next timestep's x tile (global_prefetch_b8 path).
    if (t + 1 < T_) {
        const float* px = x + (size_t)(bm0 + (tid & 31)) * (T_ * D_)
                            + (size_t)(t + 1) * D_ + (tid >> 5) * 64;
        __builtin_prefetch(px, 0, 1);
    }

    // Per-thread staging units:
    //   A: 32 rows x 64 k = 256 octets -> 2 octets/thread
    //   B: 4 gates x 32 n x 8 k-octets = 1024 octets -> 8 octets/thread
    const int aRow0 = (tid + 0)   >> 3, aOct0 = (tid + 0)   & 7;
    const int aRow1 = (tid + 128) >> 3, aOct1 = (tid + 128) & 7;

    // Ping-pong register staging buffers (kc loop fully unrolled -> SSA).
    bf16x8 aReg[2][2];
    bf16x8 bReg[2][8];

    // ---- global -> VGPR loaders ----
    auto load_a = [&](int kc, int buf) {
        const int k0 = kc * 64;
        if (kc < 4) {                    // from x_t (fp32 -> bf16)
            #pragma unroll
            for (int i = 0; i < 2; ++i) {
                int row = (i == 0) ? aRow0 : aRow1;
                int oct = (i == 0) ? aOct0 : aOct1;
                const float* p = x + (size_t)(bm0 + row) * (T_ * D_)
                                   + (size_t)t * D_ + k0 + oct * 8;
                f32x4 v0 = *(const f32x4*)(p);
                f32x4 v1 = *(const f32x4*)(p + 4);
                bf16x4 b0 = __builtin_convertvector(v0, bf16x4);
                bf16x4 b1 = __builtin_convertvector(v1, bf16x4);
                aReg[buf][i] = __builtin_shufflevector(b0, b1, 0,1,2,3,4,5,6,7);
            }
        } else {                         // from h_in (bf16)
            const int kh = k0 - D_;
            #pragma unroll
            for (int i = 0; i < 2; ++i) {
                int row = (i == 0) ? aRow0 : aRow1;
                int oct = (i == 0) ? aOct0 : aOct1;
                aReg[buf][i] = *(const bf16x8*)
                    (h_in + (size_t)(bm0 + row) * H_ + kh + oct * 8);
            }
        }
    };
    auto load_b = [&](int kc, int buf) {
        const int k0 = kc * 64;
        #pragma unroll
        for (int i = 0; i < 8; ++i) {
            int u = tid + i * 128;
            int g = u >> 8, rem = u & 255;
            int n = rem >> 3, ko = rem & 7;
            bReg[buf][i] = *(const bf16x8*)
                (Wt + (size_t)(g * H_ + bh0 + n) * (D_ + H_) + k0 + ko * 8);
        }
    };
    // ---- VGPR -> LDS ----
    auto store_stage = [&](int buf) {
        *(bf16x8*)&As[aRow0][aOct0 * 8] = aReg[buf][0];
        *(bf16x8*)&As[aRow1][aOct1 * 8] = aReg[buf][1];
        #pragma unroll
        for (int i = 0; i < 8; ++i) {
            int u = tid + i * 128;
            int g = u >> 8, rem = u & 255;
            int n = rem >> 3, ko = rem & 7;
            *(bf16x8*)&Bs[g][n][ko * 8] = bReg[buf][i];
        }
    };

    v8f acc[4] = {};                     // one per gate

    // Preload chunk 0.
    load_a(0, 0);
    load_b(0, 0);

    #pragma unroll
    for (int kc = 0; kc < 8; ++kc) {     // K = D + H = 512, chunks of 64
        const int buf = kc & 1;
        store_stage(buf);
        __syncthreads();

        // Issue next chunk's global loads now -> overlap with WMMAs below.
        if (kc < 7) {
            load_a(kc + 1, buf ^ 1);
            load_b(kc + 1, buf ^ 1);
        }

        #pragma unroll
        for (int kk = 0; kk < 64; kk += 32) {
            v16bf afrag = load_a_frag(&As[wr * 16 + (lane & 15)][0], kk + lhi * 8);
            #pragma unroll
            for (int g = 0; g < 4; ++g) {
                v16bf bfrag = load_b_frag(&Bs[g][wc * 16 + (lane & 15)][0],
                                          kk + lhi * 8);
                acc[g] = __builtin_amdgcn_wmma_f32_16x16x32_bf16(
                    false, afrag, false, bfrag, (short)0, acc[g], false, false);
            }
        }
        __syncthreads();
    }

    // ---- fused gate epilogue ----
    // C/D layout: lane L, vgpr r -> row = r + (L>=16?8:0), col = L%16
    const int colH = bh0 + wc * 16 + (lane & 15);
    const float bi = bias[0 * H_ + colH];
    const float bj = bias[1 * H_ + colH];
    const float bf = bias[2 * H_ + colH] + 1.0f;  // FORGET_BIAS fused
    const float bo = bias[3 * H_ + colH];

    #pragma unroll
    for (int r = 0; r < 8; ++r) {
        int row = bm0 + wr * 16 + r + lhi * 8;
        size_t ci = (size_t)row * H_ + colH;
        float ig = acc[0][r] + bi;
        float jg = acc[1][r] + bj;
        float fg = acc[2][r] + bf;
        float og = acc[3][r] + bo;
        float cv = c[ci];
        float nc = cv * sigf(fg) + sigf(ig) * tanhf(jg);
        float nh = tanhf(nc) * sigf(og);
        c[ci]     = nc;
        h_out[ci] = (__bf16)nh;
    }
}

// ---------------------------------------------------------------------------
// Output projection: out[B,C] = h[B,H] @ w_out[H,C] + b_out. grid (8,2),
// 256 threads (8 waves, 2x4), 64x64 tile.
// ---------------------------------------------------------------------------
__global__ void __launch_bounds__(256)
out_proj_kernel(const __bf16* __restrict__ h,      // [B,H] bf16
                const __bf16* __restrict__ WoutT,  // [C][H] bf16 (transposed)
                const float* __restrict__ b_out,   // [C]
                float* __restrict__ out) {         // [B,C]
    __shared__ __bf16 As[64][LP];
    __shared__ __bf16 Bs[64][LP];   // [n][k]

    const int tid  = threadIdx.x;
    const int lane = tid & 31;
    const int wave = tid >> 5;
    const int wr   = wave >> 2;
    const int wc   = wave & 3;
    const int bm0  = blockIdx.x * 64;
    const int cn0  = blockIdx.y * 64;
    const int lhi  = (lane & 16) ? 1 : 0;

    v8f acc[2] = {};

    for (int kc = 0; kc < 4; ++kc) {   // K = H = 256
        const int k0 = kc * 64;
        #pragma unroll
        for (int i = 0; i < 2; ++i) {
            int u = tid + i * 256;     // 512 octet units
            int row = u >> 3, o = u & 7;
            *(bf16x8*)&As[row][o * 8] =
                *(const bf16x8*)(h + (size_t)(bm0 + row) * H_ + k0 + o * 8);
        }
        #pragma unroll
        for (int i = 0; i < 2; ++i) {
            int u = tid + i * 256;
            int n = u >> 3, ko = u & 7;
            *(bf16x8*)&Bs[n][ko * 8] =
                *(const bf16x8*)(WoutT + (size_t)(cn0 + n) * H_ + k0 + ko * 8);
        }
        __syncthreads();

        #pragma unroll
        for (int kk = 0; kk < 64; kk += 32) {
            v16bf bfrag = load_b_frag(&Bs[wc * 16 + (lane & 15)][0], kk + lhi * 8);
            #pragma unroll
            for (int mt = 0; mt < 2; ++mt) {
                v16bf afrag = load_a_frag(&As[wr * 32 + mt * 16 + (lane & 15)][0],
                                          kk + lhi * 8);
                acc[mt] = __builtin_amdgcn_wmma_f32_16x16x32_bf16(
                    false, afrag, false, bfrag, (short)0, acc[mt], false, false);
            }
        }
        __syncthreads();
    }

    const int col = cn0 + wc * 16 + (lane & 15);
    const float bb = b_out[col];
    #pragma unroll
    for (int mt = 0; mt < 2; ++mt) {
        #pragma unroll
        for (int r = 0; r < 8; ++r) {
            int row = bm0 + wr * 32 + mt * 16 + r + lhi * 8;
            out[(size_t)row * C_ + col] = acc[mt][r] + bb;
        }
    }
}

// ---------------------------------------------------------------------------
extern "C" void kernel_launch(void* const* d_in, const int* in_sizes, int n_in,
                              void* d_out, int out_size, void* d_ws, size_t ws_size,
                              hipStream_t stream) {
    const float* x     = (const float*)d_in[0];  // [B,T,D]
    const float* W     = (const float*)d_in[1];  // [(D+H),4H]
    const float* bias  = (const float*)d_in[2];  // [4H]
    const float* w_out = (const float*)d_in[3];  // [H,C]
    const float* b_out = (const float*)d_in[4];  // [C]
    float* out = (float*)d_out;                  // [B,C]

    // workspace carve (~2.2 MB)
    char* ws = (char*)d_ws;
    __bf16* Wt    = (__bf16*)ws;  ws += (size_t)G4H * (D_ + H_) * sizeof(__bf16);
    __bf16* WoutT = (__bf16*)ws;  ws += (size_t)C_ * H_ * sizeof(__bf16);
    __bf16* h0    = (__bf16*)ws;  ws += (size_t)B_ * H_ * sizeof(__bf16);
    __bf16* h1    = (__bf16*)ws;  ws += (size_t)B_ * H_ * sizeof(__bf16);
    float*  c     = (float*)ws;

    lstm_prep_kernel<<<2048, 256, 0, stream>>>(W, w_out, Wt, WoutT, c, h0);

    for (int t = 0; t < T_; ++t) {
        const __bf16* hin  = (t & 1) ? h1 : h0;
        __bf16*       hout = (t & 1) ? h0 : h1;
        lstm_step_kernel<<<dim3(16, 8), 128, 0, stream>>>(
            x, Wt, bias, hin, hout, c, t);
    }
    // t = 127 (odd) wrote h0 -> final state in h0
    out_proj_kernel<<<dim3(8, 2), 256, 0, stream>>>(h0, WoutT, b_out, out);
}
